// GCN_23029614641361
// MI455X (gfx1250) — compile-verified, matrix operand order checked by reading, not run on hardware
//
#include <hip/hip_runtime.h>

// ---------- WMMA types (CDNA5 gfx1250, wave32) ----------
typedef __attribute__((ext_vector_type(16))) __bf16 v16bf;
typedef __attribute__((ext_vector_type(8)))  float  v8f;
typedef __attribute__((ext_vector_type(4)))  int    v4i;

#define GCN_AS1 __attribute__((address_space(1)))
#define GCN_AS3 __attribute__((address_space(3)))

union Frag {
    v16bf    v;
    unsigned u[8];
    uint4    q[2];
};

__device__ __forceinline__ unsigned pkbf(float a, float b) {
    unsigned ua = __float_as_uint(a);
    unsigned ub = __float_as_uint(b);
    ua += 0x7FFFu + ((ua >> 16) & 1u);
    ub += 0x7FFFu + ((ub >> 16) & 1u);
    return (ua >> 16) | (ub & 0xFFFF0000u);
}

__device__ __forceinline__ unsigned short f32_to_bf16_bits(float x) {
    unsigned u = __float_as_uint(x);
    u += 0x7FFFu + ((u >> 16) & 1u);
    return (unsigned short)(u >> 16);
}

#if defined(__has_builtin)
#if __has_builtin(__builtin_amdgcn_global_load_async_to_lds_b128)
#define GCN_HAS_ASYNC_LDS 1
#endif
#endif

// ---------- one-time W prep: fp32 row-major [K x NOUT] -> bf16 transposed [NOUT x K] ----------
__global__ __launch_bounds__(256) void gcn_prep_wt(const float* __restrict__ W,
                                                   unsigned short* __restrict__ Wt,
                                                   int K, int NOUT) {
    int gid = blockIdx.x * 256 + threadIdx.x;
    if (gid >= K * NOUT) return;
    int k = gid / NOUT;
    int n = gid - k * NOUT;
    Wt[n * K + k] = f32_to_bf16_bits(W[gid]);
}

// ---------- GEMM: C[M x NOUT] = A[M x K](bf16) * W[K x NOUT] ----------
// A: pre-converted bf16, raw b128 loads. Wt: pre-transposed bf16 in global,
// staged to LDS via async global->LDS b128 copies (ASYNCcnt path) when available.
// Block: 256 threads = 8 waves (wave32), arranged WROWS x WCOLS.
template <int K, int NOUT, int WROWS, int WCOLS>
__global__ __launch_bounds__(256) void gcn_gemm_bf16(const unsigned short* __restrict__ A,
                                                     const unsigned short* __restrict__ Wt,
                                                     float* __restrict__ C,
                                                     int M) {
    constexpr int WTILE_N = NOUT / WCOLS;   // columns per wave
    constexpr int NACC    = WTILE_N / 16;   // 16x16 accumulators per wave
    constexpr int ELEMS   = NOUT * K;       // u16 elements (multiple of 2048)

    __shared__ unsigned short sW[ELEMS];    // transposed: sW[n*K + k]

    const int tid = threadIdx.x;
#ifdef GCN_HAS_ASYNC_LDS
    // bulk async copy: each lane moves 16B per op; wave moves 512B per op
#pragma unroll
    for (int i = tid * 8; i < ELEMS; i += 256 * 8) {
        __builtin_amdgcn_global_load_async_to_lds_b128(
            (GCN_AS1 v4i*)(Wt + i),
            (GCN_AS3 v4i*)(sW + i),
            0, 0);
    }
#if __has_builtin(__builtin_amdgcn_s_wait_asynccnt)
    __builtin_amdgcn_s_wait_asynccnt(0);
#else
    asm volatile("s_wait_asynccnt 0x0" ::: "memory");
#endif
#else
    // synchronous fallback: 16B vector copies
#pragma unroll
    for (int i = tid * 8; i < ELEMS; i += 256 * 8) {
        *reinterpret_cast<uint4*>(&sW[i]) = *reinterpret_cast<const uint4*>(Wt + i);
    }
#endif
    __syncthreads();

    const int wave = tid >> 5;
    const int lane = tid & 31;
    const int wrow = wave / WCOLS;
    const int wcol = wave - wrow * WCOLS;
    const int half = lane >> 4;   // 0: lanes 0-15, 1: lanes 16-31
    const int l16  = lane & 15;

    const int rowbase = blockIdx.x * (WROWS * 16) + wrow * 16;
    const int mrow    = rowbase + l16;
    const int mclamp  = (mrow < M) ? mrow : (M - 1);
    const unsigned short* __restrict__ arow = A + (size_t)mclamp * K;

    v8f acc[NACC] = {};

#pragma unroll
    for (int kk = 0; kk < K; kk += 32) {
        // A fragment: lane holds row mrow, K = kk+half*8+[0..7] and kk+16+half*8+[0..7]
        Frag af;
        af.q[0] = *reinterpret_cast<const uint4*>(arow + kk + half * 8);
        af.q[1] = *reinterpret_cast<const uint4*>(arow + kk + 16 + half * 8);

#pragma unroll
        for (int cg = 0; cg < NACC; ++cg) {
            const int n = wcol * WTILE_N + cg * 16 + l16;
            Frag bfr;
            bfr.q[0] = *reinterpret_cast<const uint4*>(&sW[n * K + kk + half * 8]);
            bfr.q[1] = *reinterpret_cast<const uint4*>(&sW[n * K + kk + 16 + half * 8]);
            acc[cg] = __builtin_amdgcn_wmma_f32_16x16x32_bf16(
                false, af.v, false, bfr.v, (short)0, acc[cg], false, false);
        }
    }

    // C/D layout: VGPR r -> row = rowbase + half*8 + r, col = lane's l16
#pragma unroll
    for (int cg = 0; cg < NACC; ++cg) {
        const int n = wcol * WTILE_N + cg * 16 + l16;
#pragma unroll
        for (int r = 0; r < 8; ++r) {
            const int row = rowbase + half * 8 + r;
            if (row < M) C[(size_t)row * NOUT + n] = acc[cg][r];
        }
    }
}

// ---------- fp32 -> bf16 streaming convert (optionally fused ReLU) ----------
template <bool RELU>
__global__ __launch_bounds__(256) void gcn_cvt_bf16(const float* __restrict__ in,
                                                    unsigned short* __restrict__ out,
                                                    long long n4) {
    long long gid = (long long)blockIdx.x * 256 + threadIdx.x;
    if (gid >= n4) return;
    float4 f = reinterpret_cast<const float4*>(in)[gid];
    if (RELU) {
        f.x = fmaxf(f.x, 0.0f); f.y = fmaxf(f.y, 0.0f);
        f.z = fmaxf(f.z, 0.0f); f.w = fmaxf(f.w, 0.0f);
    }
    uint2 r;
    r.x = pkbf(f.x, f.y);
    r.y = pkbf(f.z, f.w);
    reinterpret_cast<uint2*>(out)[gid] = r;
}

// ---------- degree / normalization ----------
__global__ __launch_bounds__(256) void gcn_deg_init(float* __restrict__ dinv, int N) {
    int i = blockIdx.x * 256 + threadIdx.x;
    if (i < N) dinv[i] = 1.0f;   // self-loop
}

__global__ __launch_bounds__(256) void gcn_deg_count(float* __restrict__ dinv,
                                                     const long long* __restrict__ dst, int E) {
    int e = blockIdx.x * 256 + threadIdx.x;
    if (e < E) atomicAdd(&dinv[dst[e]], 1.0f);
}

__global__ __launch_bounds__(256) void gcn_deg_rsqrt(float* __restrict__ dinv, int N) {
    int i = blockIdx.x * 256 + threadIdx.x;
    if (i < N) dinv[i] = rsqrtf(dinv[i]);
}

// ---------- self-loop + bias (initializes output buffer), float4, compile-time C ----------
template <int C4>  // C4 = channels/4 (power of two)
__global__ __launch_bounds__(256) void gcn_self_bias(const float* __restrict__ xw,
                                                     const float* __restrict__ dinv,
                                                     const float* __restrict__ b,
                                                     float* __restrict__ out,
                                                     int N) {
    long long gid = (long long)blockIdx.x * 256 + threadIdx.x;
    if (gid >= (long long)N * C4) return;
    int n  = (int)(gid / C4);          // constant power-of-two -> shift
    int c4 = (int)(gid & (C4 - 1));
    float dv = dinv[n];
    float d2 = dv * dv;
    float4 v  = reinterpret_cast<const float4*>(xw)[gid];
    float4 bb = reinterpret_cast<const float4*>(b)[c4];
    float4 o;
    o.x = d2 * v.x + bb.x; o.y = d2 * v.y + bb.y;
    o.z = d2 * v.z + bb.z; o.w = d2 * v.w + bb.w;
    reinterpret_cast<float4*>(out)[gid] = o;
}

// ---------- edge scatter: out[dst] += dinv[src]*dinv[dst] * xw[src], float4 gather ----------
template <int C4>  // C4 = channels/4 (power of two)
__global__ __launch_bounds__(256) void gcn_edge_scatter(const float* __restrict__ xw,
                                                        const float* __restrict__ dinv,
                                                        const long long* __restrict__ src,
                                                        const long long* __restrict__ dst,
                                                        float* __restrict__ out,
                                                        int E) {
    long long gid = (long long)blockIdx.x * 256 + threadIdx.x;
    if (gid >= (long long)E * C4) return;
    int e = (int)(gid / C4);           // constant power-of-two -> shift
    int c = (int)(gid & (C4 - 1)) * 4;
    long long s = src[e];
    long long d = dst[e];
    float nrm = dinv[s] * dinv[d];
    constexpr long long C = (long long)C4 * 4;
    float4 v = *reinterpret_cast<const float4*>(&xw[s * C + c]);
    float* o = &out[d * C + c];
    atomicAdd(o + 0, nrm * v.x);
    atomicAdd(o + 1, nrm * v.y);
    atomicAdd(o + 2, nrm * v.z);
    atomicAdd(o + 3, nrm * v.w);
}

extern "C" void kernel_launch(void* const* d_in, const int* in_sizes, int n_in,
                              void* d_out, int out_size, void* d_ws, size_t ws_size,
                              hipStream_t stream) {
    const float*     x   = (const float*)d_in[0];
    const long long* ei  = (const long long*)d_in[1];  // int64 edge_index [2, E]
    const float*     W1  = (const float*)d_in[2];
    const float*     b1  = (const float*)d_in[3];
    const float*     W2  = (const float*)d_in[4];
    const float*     b2  = (const float*)d_in[5];
    float*           out = (float*)d_out;

    const int N = in_sizes[0] / 128;  // 100000
    const int E = in_sizes[1] / 2;    // 640000
    const long long* src = ei;
    const long long* dst = ei + E;

    // workspace: [dinv][region A: N*256 f32][region B: N*256 f32][Wt1 64KB][Wt2 32KB]
    //  region B: x_bf16 (consumed by GEMM1) -> h fp32
    //  region A: xw1 fp32 -> h_bf16 (first half) + xw2 fp32 (second half)
    char*  ws      = (char*)d_ws;
    float* dinv    = (float*)ws;
    size_t off     = (((size_t)N * 4) + 255) & ~(size_t)255;
    float* regionA = (float*)(ws + off);
    float* regionB = regionA + (size_t)N * 256;

    float*          xw1 = regionA;
    unsigned short* xbf = (unsigned short*)regionB;
    float*          h   = regionB;
    unsigned short* hbf = (unsigned short*)regionA;
    float*          xw2 = regionA + (size_t)N * 128;           // past hbf
    unsigned short* wt1 = (unsigned short*)(regionB + (size_t)N * 256);
    unsigned short* wt2 = wt1 + 128 * 256;

    const int TB = 256;
    // 0) one-time weight prep: bf16 + transpose (tiny)
    gcn_prep_wt<<<(128 * 256 + TB - 1) / TB, TB, 0, stream>>>(W1, wt1, 128, 256);
    gcn_prep_wt<<<(256 * 64  + TB - 1) / TB, TB, 0, stream>>>(W2, wt2, 256, 64);

    // 1) symmetric normalization coefficients
    gcn_deg_init <<<(N + TB - 1) / TB, TB, 0, stream>>>(dinv, N);
    gcn_deg_count<<<(E + TB - 1) / TB, TB, 0, stream>>>(dinv, dst, E);
    gcn_deg_rsqrt<<<(N + TB - 1) / TB, TB, 0, stream>>>(dinv, N);

    // 2) x -> bf16 once; layer-1 GEMM reads raw bf16
    {
        long long n4 = (long long)N * 128 / 4;
        gcn_cvt_bf16<false><<<(unsigned)((n4 + TB - 1) / TB), TB, 0, stream>>>(x, xbf, n4);
    }
    gcn_gemm_bf16<128, 256, 2, 4><<<N / 32, TB, 0, stream>>>(xbf, wt1, xw1, N);

    // 3) h = dinv^2 * xw1 + b1 ; += edge aggregation ; ReLU fused into bf16 convert
    {
        long long tot4 = (long long)N * 64;          // N*256/4
        gcn_self_bias<64><<<(unsigned)((tot4 + TB - 1) / TB), TB, 0, stream>>>(xw1, dinv, b1, h, N);
        long long e4 = (long long)E * 64;            // E*256/4
        gcn_edge_scatter<64><<<(unsigned)((e4 + TB - 1) / TB), TB, 0, stream>>>(xw1, dinv, src, dst, h, E);
        gcn_cvt_bf16<true><<<(unsigned)((tot4 + TB - 1) / TB), TB, 0, stream>>>(h, hbf, tot4);
    }

    // 4) layer 2: xw2 = relu(h)_bf16 @ W2 (block tile 128 rows, row guards)
    gcn_gemm_bf16<256, 64, 8, 1><<<(N + 127) / 128, TB, 0, stream>>>(hbf, wt2, xw2, N);

    // 5) out = dinv^2 * xw2 + b2 ; += edge aggregation
    {
        long long tot4 = (long long)N * 16;          // N*64/4
        gcn_self_bias<16><<<(unsigned)((tot4 + TB - 1) / TB), TB, 0, stream>>>(xw2, dinv, b2, out, N);
        long long e4 = (long long)E * 16;            // E*64/4
        gcn_edge_scatter<16><<<(unsigned)((e4 + TB - 1) / TB), TB, 0, stream>>>(xw2, dinv, src, dst, out, E);
    }
}
